// ViziMarketTransformer_67095979098454
// MI455X (gfx1250) — matrix operand order repaired and checked by reading, not compiled, not run on hardware
//
#include <hip/hip_runtime.h>
#include <hip/hip_bf16.h>
#include <math.h>

// MoE FFN: D=512, F=2048, E=8, top-2, N = 32768 tokens. fp32 in/out.
// Strategy: weights pre-swizzled to bf16 in WMMA B-fragment order (once per
// launch), fused GEMM1 -> GELU -> comb-scale -> GEMM2 per 32-token tile with
// bf16 WMMA / f32 accumulation. h intermediate never touches HBM.

#define DMODEL 512
#define DFF    2048
#define NEXP   8
#define NTOK   32768
#define MT     32           // tokens per workgroup
#define FC     128          // F-chunk width (8 waves x 16 cols)
#define XLD    (DMODEL + 8) // padded LDS strides (bank-conflict avoidance)
#define HLD    (FC + 8)
#define KT1    (DMODEL / 32)  // 16  k-tiles of w1 per expert
#define NT1    (DFF / 16)     // 128 n-tiles of w1 per expert
#define KT2    (DFF / 32)     // 64  k-tiles of w2 per expert
#define NT2    (DMODEL / 16)  // 32  n-tiles of w2 per expert

typedef __bf16 v16bf __attribute__((ext_vector_type(16)));
typedef __bf16 v8bf  __attribute__((ext_vector_type(8)));
typedef float  v8f   __attribute__((ext_vector_type(8)));

__device__ __forceinline__ v8f wmma_bf16(v16bf a, v16bf b, v8f c) {
  return __builtin_amdgcn_wmma_f32_16x16x32_bf16(false, a, false, b, (short)0, c,
                                                 false, false);
}

// A (16x32 bf16, row-major LDS): lane m = lane&15, koff = (lane>>4)*16.
// Frag elems 0..7 = K koff..koff+7; elems 8..15 = K koff+8..koff+15:
// two contiguous 16B runs -> 2x ds_load_b128.
__device__ __forceinline__ v16bf load_a_lds(const __bf16* base, int lda, int lane) {
  int m    = lane & 15;
  int koff = (lane >> 4) * 16;
  v8bf r0 = *(const v8bf*)(base + m * lda + koff);
  v8bf r1 = *(const v8bf*)(base + m * lda + koff + 8);
  v16bf a;
#pragma unroll
  for (int j = 0; j < 8; ++j) { a[j] = r0[j]; a[8 + j] = r1[j]; }
  return a;
}

// B fragment from pre-swizzled weights: 32 contiguous bytes per lane,
// one wave-load = 1KB contiguous.
__device__ __forceinline__ v16bf load_b_swz(const __bf16* tile, int lane) {
  return *(const v16bf*)(tile + lane * 16);
}

// Branchless exact-GELU (Abramowitz-Stegun 7.1.26 erf, |err| ~1.5e-7).
// Uses hardware v_rcp_f32 instead of the IEEE division sequence.
__device__ __forceinline__ float gelu_exact(float x) {
  float z = x * 0.70710678118654752f;
  float a = fabsf(z);
  float t = __builtin_amdgcn_rcpf(1.0f + 0.3275911f * a);  // single v_rcp_f32
  float y = t * (0.254829592f +
            t * (-0.284496736f +
            t * (1.421413741f +
            t * (-1.453152027f + t * 1.061405429f))));
  float erfa = 1.0f - y * __expf(-z * z);
  return 0.5f * x * (1.0f + copysignf(erfa, z));
}

// ---------------- weight swizzle: fp32 row-major -> bf16 B-frag tiles -------
// For W (per expert: Ktot x Ntot, row-major), tile (kt,nt) is 32x16.
// dst[tile*512 + lane*16 + j] = W[(kt*32 + (lane>>4)*16 + j)*Ntot + nt*16 + (lane&15)]
__global__ __launch_bounds__(256) void moe_swizzle_kernel(
    const float* __restrict__ src, __bf16* __restrict__ dst, int NT, int Ntot) {
  size_t gid  = (size_t)blockIdx.x * 256 + threadIdx.x;
  size_t tile = gid >> 5;
  int    lane = (int)(gid & 31);
  size_t nt   = tile % NT;
  size_t ekt  = tile / NT;  // e*KT + kt  (expert matrices are row-contiguous)
  const float* s = src + (ekt * 32 + (size_t)((lane >> 4) * 16)) * Ntot +
                   nt * 16 + (lane & 15);
  __bf16* d = dst + tile * 512 + (size_t)lane * 16;
#pragma unroll
  for (int j = 0; j < 16; ++j) d[j] = (__bf16)s[(size_t)j * Ntot];
}

// ------------------------------- router -------------------------------------
__global__ __launch_bounds__(256) void moe_router_kernel(
    const float* __restrict__ x, const float* __restrict__ rw,
    float* __restrict__ comb, float* __restrict__ sums) {
  int wave = threadIdx.x >> 5;
  int lane = threadIdx.x & 31;
  int n    = blockIdx.x * 8 + wave;
  const float* xr = x + (size_t)n * DMODEL;

  float acc[NEXP];
#pragma unroll
  for (int e = 0; e < NEXP; ++e) acc[e] = 0.f;
  for (int j = 0; j < DMODEL / 32; ++j) {
    int   d  = lane + 32 * j;
    float xv = xr[d];
#pragma unroll
    for (int e = 0; e < NEXP; ++e) acc[e] += xv * rw[d * NEXP + e];
  }
#pragma unroll
  for (int off = 16; off >= 1; off >>= 1) {
#pragma unroll
    for (int e = 0; e < NEXP; ++e) acc[e] += __shfl_xor(acc[e], off, 32);
  }
  if (lane == 0) {
    int i0 = 0; float v0 = acc[0];
#pragma unroll
    for (int e = 1; e < NEXP; ++e) if (acc[e] > v0) { v0 = acc[e]; i0 = e; }
    int i1 = -1; float v1 = -3.4e38f;
#pragma unroll
    for (int e = 0; e < NEXP; ++e)
      if (e != i0 && acc[e] > v1) { v1 = acc[e]; i1 = e; }

    float e1 = __expf(v1 - v0);
    float w0 = 1.0f / (1.0f + e1);
    float w1 = e1 / (1.0f + e1);
    float c[NEXP];
#pragma unroll
    for (int e = 0; e < NEXP; ++e) c[e] = 0.f;
    c[i0] = w0; c[i1] = w1;
#pragma unroll
    for (int e = 0; e < NEXP; ++e) comb[(size_t)n * NEXP + e] = c[e];

    float p[NEXP], s = 0.f;
#pragma unroll
    for (int e = 0; e < NEXP; ++e) { p[e] = __expf(acc[e] - v0); s += p[e]; }
    float inv = 1.0f / s;
#pragma unroll
    for (int e = 0; e < NEXP; ++e) atomicAdd(&sums[e], p[e] * inv);
    atomicAdd(&sums[NEXP + i0], 1.0f);
  }
}

// ------------------------------ fused FFN -----------------------------------
__global__ __launch_bounds__(256) void moe_ffn_kernel(
    const float* __restrict__ x, const __bf16* __restrict__ w1s,
    const float* __restrict__ b1, const __bf16* __restrict__ w2s,
    const float* __restrict__ b2, const float* __restrict__ comb,
    float* __restrict__ out) {
  __shared__ __bf16 x_lds[MT * XLD];     // ~33 KB
  __shared__ __bf16 h_lds[MT * HLD];     // ~8.7 KB
  __shared__ float  comb_lds[MT * NEXP]; // 1 KB
  __shared__ float  emax[NEXP];

  int wave = threadIdx.x >> 5;
  int lane = threadIdx.x & 31;
  int row0 = blockIdx.x * MT;

  // stage x tile (fp32 -> bf16) with float4 loads
#pragma unroll
  for (int j = 0; j < (MT * DMODEL / 4) / 256; ++j) {
    int i4 = threadIdx.x + j * 256;
    int r  = (i4 * 4) >> 9, c = (i4 * 4) & (DMODEL - 1);
    float4 v = ((const float4*)x)[((size_t)(row0 + r) * DMODEL + c) >> 2];
    __bf16* dp = &x_lds[r * XLD + c];
    dp[0] = (__bf16)v.x; dp[1] = (__bf16)v.y;
    dp[2] = (__bf16)v.z; dp[3] = (__bf16)v.w;
  }
  if (threadIdx.x < MT * NEXP)
    comb_lds[threadIdx.x] = comb[(size_t)row0 * NEXP + threadIdx.x];
  __syncthreads();
  if (threadIdx.x < NEXP) {
    float m = 0.f;
    for (int r = 0; r < MT; ++r) m = fmaxf(m, comb_lds[r * NEXP + threadIdx.x]);
    emax[threadIdx.x] = m;
  }
  __syncthreads();

  v8f acc[2][4];  // [m-row 0/1][n-tile]; wave owns out cols [wave*64, +64)
#pragma unroll
  for (int mr = 0; mr < 2; ++mr)
#pragma unroll
    for (int t = 0; t < 4; ++t) acc[mr][t] = (v8f)(0.f);

  for (int e = 0; e < NEXP; ++e) {
    if (emax[e] == 0.f) continue;  // block-uniform skip
    const __bf16* w1e = w1s + (size_t)e * KT1 * NT1 * 512;
    const __bf16* w2e = w2s + (size_t)e * KT2 * NT2 * 512;

    // hoist this lane's per-row comb weights for expert e (invariant over fc)
    float cw[16];
#pragma unroll
    for (int v = 0; v < 8; ++v) {
      int m     = v + ((lane >> 4) << 3);
      cw[v]     = comb_lds[m * NEXP + e];
      cw[v + 8] = comb_lds[(m + 16) * NEXP + e];
    }

    for (int fc = 0; fc < DFF; fc += FC) {
      // ---- GEMM1: wave computes h tiles (m=0,1) x col 'wave', K = 512 ----
      v8f h0 = (v8f)(0.f), h1 = (v8f)(0.f);
      int nt1 = (fc >> 4) + wave;
#pragma unroll 2
      for (int kk = 0; kk < DMODEL; kk += 32) {
        int   kt = kk >> 5;
        v16bf b  = load_b_swz(w1e + (size_t)(kt * NT1 + nt1) * 512, lane);
        v16bf a0 = load_a_lds(x_lds + kk, XLD, lane);
        v16bf a1 = load_a_lds(x_lds + 16 * XLD + kk, XLD, lane);
        h0 = wmma_bf16(a0, b, h0);
        h1 = wmma_bf16(a1, b, h1);
      }
      // epilogue: b1 + exact GELU + per-row comb scale -> bf16 LDS
      {
        int   n    = lane & 15;
        float bias = b1[e * DFF + fc + wave * 16 + n];
#pragma unroll
        for (int v = 0; v < 8; ++v) {
          int m = v + ((lane >> 4) << 3);
          h_lds[m * HLD + wave * 16 + n] =
              (__bf16)(gelu_exact(h0[v] + bias) * cw[v]);
          h_lds[(m + 16) * HLD + wave * 16 + n] =
              (__bf16)(gelu_exact(h1[v] + bias) * cw[v + 8]);
        }
      }
      __syncthreads();

      // ---- GEMM2: out[32, wave*64..+64) += h(32x128) @ w2 chunk ----
#pragma unroll 2
      for (int kk = 0; kk < FC; kk += 32) {
        int   kt2 = (fc + kk) >> 5;
        v16bf a0  = load_a_lds(h_lds + kk, HLD, lane);
        v16bf a1  = load_a_lds(h_lds + 16 * HLD + kk, HLD, lane);
#pragma unroll
        for (int t = 0; t < 4; ++t) {
          int   nt2 = wave * 4 + t;
          v16bf b   = load_b_swz(w2e + (size_t)(kt2 * NT2 + nt2) * 512, lane);
          acc[0][t] = wmma_bf16(a0, b, acc[0][t]);
          acc[1][t] = wmma_bf16(a1, b, acc[1][t]);
        }
      }
      __syncthreads();  // protect h_lds before next chunk rewrites it
    }
  }

  // epilogue: add sum_e comb[m,e]*b2[e,n], write fp32 out
  {
    int n = lane & 15;
#pragma unroll
    for (int t = 0; t < 4; ++t) {
      int ng = wave * 64 + t * 16 + n;
#pragma unroll
      for (int mr = 0; mr < 2; ++mr) {
#pragma unroll
        for (int v = 0; v < 8; ++v) {
          int   m  = v + ((lane >> 4) << 3) + mr * 16;
          float bb = 0.f;
#pragma unroll
          for (int e = 0; e < NEXP; ++e)
            bb += comb_lds[m * NEXP + e] * b2[e * DMODEL + ng];
          out[(size_t)(row0 + m) * DMODEL + ng] = acc[mr][t][v] + bb;
        }
      }
    }
  }
}

// ------------------------------ aux / init ----------------------------------
__global__ void moe_zero_sums_kernel(float* p) {
  if (threadIdx.x < 2 * NEXP) p[threadIdx.x] = 0.f;
}

__global__ void moe_aux_kernel(const float* __restrict__ sums,
                               float* __restrict__ o, float invN) {
  if (threadIdx.x == 0) {
    float a = 0.f;
#pragma unroll
    for (int e = 0; e < NEXP; ++e)
      a += (sums[e] * invN) * (sums[NEXP + e] * invN);
    *o = 0.01f * (float)NEXP * a;
  }
}

// ------------------------------- launcher -----------------------------------
extern "C" void kernel_launch(void* const* d_in, const int* in_sizes, int n_in,
                              void* d_out, int out_size, void* d_ws, size_t ws_size,
                              hipStream_t stream) {
  (void)in_sizes; (void)n_in; (void)out_size; (void)ws_size;
  const float* x  = (const float*)d_in[0];
  const float* rw = (const float*)d_in[1];
  const float* w1 = (const float*)d_in[2];
  const float* b1 = (const float*)d_in[3];
  const float* w2 = (const float*)d_in[4];
  const float* b2 = (const float*)d_in[5];
  float* out = (float*)d_out;

  float*  ws   = (float*)d_ws;
  float*  sums = ws;                       // 16 floats
  float*  comb = ws + 16;                  // NTOK*NEXP floats (1 MB)
  __bf16* w1s  = (__bf16*)(ws + 16 + (size_t)NTOK * NEXP);  // 16.8 MB, 32B-aligned
  __bf16* w2s  = w1s + (size_t)NEXP * DMODEL * DFF;         // 16.8 MB

  moe_zero_sums_kernel<<<1, 32, 0, stream>>>(sums);
  moe_router_kernel<<<NTOK / 8, 256, 0, stream>>>(x, rw, comb, sums);
  // swizzle weights: one thread per (tile, lane); exact grids
  moe_swizzle_kernel<<<(NEXP * KT1 * NT1 * 32) / 256, 256, 0, stream>>>(w1, w1s, NT1, DFF);
  moe_swizzle_kernel<<<(NEXP * KT2 * NT2 * 32) / 256, 256, 0, stream>>>(w2, w2s, NT2, DMODEL);
  moe_ffn_kernel<<<NTOK / MT, 256, 0, stream>>>(x, w1s, b1, w2s, b2, comb, out);
  moe_aux_kernel<<<1, 32, 0, stream>>>(sums, out + (size_t)NTOK * DMODEL,
                                       1.0f / (float)NTOK);
}
